// RQVAE_82970178224790
// MI455X (gfx1250) — matrix operand order repaired and checked by reading, not compiled, not run on hardware
//
#include <hip/hip_runtime.h>

#define NROWS 65536
#define INDIM 768
#define EMB   128

typedef __attribute__((ext_vector_type(16))) __bf16          v16bf;
typedef __attribute__((ext_vector_type(16))) unsigned short  v16u;
typedef __attribute__((ext_vector_type(8)))  float           v8f;
typedef __attribute__((ext_vector_type(4)))  unsigned int    v4u;
typedef __attribute__((ext_vector_type(8)))  int             v8i;
typedef __attribute__((ext_vector_type(4)))  int             v4i;

#if __has_builtin(__builtin_amdgcn_tensor_load_to_lds) && \
    __has_builtin(__builtin_amdgcn_s_wait_tensorcnt)
#define HAVE_TDM 1
#else
#define HAVE_TDM 0
#endif

__device__ __forceinline__ unsigned short f2bf(float f) {
  return __builtin_bit_cast(unsigned short, (__bf16)f);   // native v_cvt, RNE
}

// A-fragment (16x32 bf16), CDNA5 ISA layout:
//  lanes 0-15 : slots0-7 = K0-7,  slots8-15 = K16-23 (row M = lane)
//  lanes16-31 : slots0-7 = K8-15, slots8-15 = K24-31 (row M = lane-16)
__device__ __forceinline__ v16bf a_frag_f32(const float* p, int kh) {
  v16bf u;
#pragma unroll
  for (int i = 0; i < 8; ++i) u[i]     = (__bf16)p[kh * 8 + i];
#pragma unroll
  for (int i = 0; i < 8; ++i) u[8 + i] = (__bf16)p[16 + kh * 8 + i];
  return u;
}
__device__ __forceinline__ v16bf a_frag_bf16(const unsigned short* p, int kh) {
  v16u u;
#pragma unroll
  for (int i = 0; i < 8; ++i) u[i]     = p[kh * 8 + i];
#pragma unroll
  for (int i = 0; i < 8; ++i) u[8 + i] = p[16 + kh * 8 + i];
  return __builtin_bit_cast(v16bf, u);
}
// B-fragment (32x16 bf16): column n of B = row n of W (contiguous over K)
//  lanes 0-15 : K0-7,K16-23 (col n = lane), lanes16-31 : K8-15,K24-31
__device__ __forceinline__ v16bf b_frag_f32(const float* p, int kh) {
  v16bf u;
#pragma unroll
  for (int i = 0; i < 16; ++i) u[i] = (__bf16)p[(i < 8 ? 0 : 8) + kh * 8 + i];
  return u;
}
__device__ __forceinline__ v16bf b_frag_bf16(const unsigned short* p, int kh) {
  v16u u;
#pragma unroll
  for (int i = 0; i < 16; ++i) u[i] = p[(i < 8 ? 0 : 8) + kh * 8 + i];
  return __builtin_bit_cast(v16bf, u);
}

__device__ __forceinline__ v8f wmma_bf16(v16bf a, v16bf b, v8f c) {
  return __builtin_amdgcn_wmma_f32_16x16x32_bf16(false, a, false, b,
                                                 (short)0, c, false, false);
}

#if HAVE_TDM
__device__ __forceinline__ unsigned lds_addr_of(void* p) {
  return (unsigned)(unsigned long long)(__attribute__((address_space(3))) void*)p;
}
// 2D TDM load: tile (tile0 x tile1) of f32 tensor (dim0 x dim1), row pitch
// stride0 (elements), into LDS at byte offset lds_addr.  D# per ISA 08 §8.
// Toolchain uses the 6-arg builtin: (v4u g0, v8i g1, v4i g2, v4i g3, v8i, cpol)
__device__ __forceinline__ void tdm_load_2d(unsigned lds_addr, const void* gptr,
                                            unsigned dim0, unsigned dim1,
                                            unsigned tile0, unsigned tile1,
                                            unsigned long long stride0) {
  unsigned long long ga = (unsigned long long)gptr;
  v4u g0;
  g0[0] = 1u;                                       // count=1, user descriptor
  g0[1] = lds_addr;                                 // bits 63:32  lds_addr
  g0[2] = (unsigned)ga;                             // bits 95:64  global_addr lo
  g0[3] = (unsigned)((ga >> 32) & 0x01FFFFFFu)      // bits 120:96 global_addr hi
        | 0x80000000u;                              // bits 127:126 type=2
  v8i g1;
  g1[0] = (int)(2u << 16);                          // data_size=4B, no mask/pad
  g1[1] = (int)((dim0 & 0xFFFFu) << 16);            // tensor_dim0[15:0]
  g1[2] = (int)((dim0 >> 16) | ((dim1 & 0xFFFFu) << 16));
  g1[3] = (int)((dim1 >> 16) | (tile0 << 16));      // tile_dim0
  g1[4] = (int)(tile1 & 0xFFFFu);                   // tile_dim1, tile_dim2=0
  g1[5] = (int)(unsigned)(stride0 & 0xFFFFFFFFull); // tensor_dim0_stride lo
  g1[6] = (int)(unsigned)(stride0 >> 32);           // stride hi, dim1_stride=0
  g1[7] = 0;
  v4i z4 = {0, 0, 0, 0};
  v8i z8 = {0, 0, 0, 0, 0, 0, 0, 0};
  __builtin_amdgcn_tensor_load_to_lds(g0, g1, z4, z4, z8, 0);
}
#endif

// ---------------- encoder layer 1: h = relu(x @ W_e1^T + b), h stored bf16 --
__global__ __launch_bounds__(256) void enc_gemm1(const float* __restrict__ x,
                                                 const float* __restrict__ W,
                                                 const float* __restrict__ b,
                                                 unsigned short* __restrict__ h) {
  __shared__ float Bt[2][64 * 32];                  // 16 KB double-buffered tile
  const int lane = threadIdx.x & 31;
  const int wave = threadIdx.x >> 5;
  const int kh   = lane >> 4;
  const int ln   = lane & 15;
  const size_t m0 = (size_t)blockIdx.x * 128 + wave * 16;
  const int    n0 = blockIdx.y * 64;
  const float* arow = x + (m0 + ln) * INDIM;
  const int nit = INDIM / 32;

#if HAVE_TDM
  if (wave == 0)
    tdm_load_2d(lds_addr_of(&Bt[0][0]), W + (size_t)n0 * INDIM,
                INDIM, INDIM, 32, 64, INDIM);
#endif
  v8f acc[4] = {};
  for (int it = 0; it < nit; ++it) {
    const int cur = it & 1;
    const int k0 = it * 32;
#if HAVE_TDM
    if (wave == 0) __builtin_amdgcn_s_wait_tensorcnt(0);
    __syncthreads();                                // tile[cur] ready, prev retired
    if (wave == 0 && it + 1 < nit)
      tdm_load_2d(lds_addr_of(&Bt[cur ^ 1][0]),
                  W + (size_t)n0 * INDIM + (it + 1) * 32,
                  INDIM, INDIM, 32, 64, INDIM);
#else
    __syncthreads();
    for (int t = threadIdx.x; t < 64 * 32; t += 256)
      Bt[cur][t] = W[(size_t)(n0 + (t >> 5)) * INDIM + k0 + (t & 31)];
    __syncthreads();
#endif
    if (k0 + 32 < INDIM) __builtin_prefetch(arow + k0 + 32, 0, 0);
    v16bf a = a_frag_f32(arow + k0, kh);
#pragma unroll
    for (int j = 0; j < 4; ++j) {
      v16bf bf = b_frag_f32(&Bt[cur][(j * 16 + ln) * 32], kh);
      acc[j] = wmma_bf16(a, bf, acc[j]);
    }
  }
#pragma unroll
  for (int j = 0; j < 4; ++j) {
    int n = n0 + j * 16 + ln;
    float bias = b[n];
#pragma unroll
    for (int v = 0; v < 8; ++v) {
      size_t m = m0 + kh * 8 + v;
      float val = acc[j][v] + bias;
      h[m * INDIM + n] = f2bf(val > 0.f ? val : 0.f);
    }
  }
}

// ---------------- encoder layer 2: z = h(bf16) @ W_e2^T + b (f32 out) -------
__global__ __launch_bounds__(256) void enc_gemm2(const unsigned short* __restrict__ h,
                                                 const float* __restrict__ W,
                                                 const float* __restrict__ b,
                                                 float* __restrict__ z) {
  __shared__ float Bt[2][64 * 32];
  const int lane = threadIdx.x & 31;
  const int wave = threadIdx.x >> 5;
  const int kh   = lane >> 4;
  const int ln   = lane & 15;
  const size_t m0 = (size_t)blockIdx.x * 128 + wave * 16;
  const int    n0 = blockIdx.y * 64;
  const unsigned short* arow = h + (m0 + ln) * INDIM;
  const int nit = INDIM / 32;

#if HAVE_TDM
  if (wave == 0)
    tdm_load_2d(lds_addr_of(&Bt[0][0]), W + (size_t)n0 * INDIM,
                INDIM, EMB, 32, 64, INDIM);
#endif
  v8f acc[4] = {};
  for (int it = 0; it < nit; ++it) {
    const int cur = it & 1;
    const int k0 = it * 32;
#if HAVE_TDM
    if (wave == 0) __builtin_amdgcn_s_wait_tensorcnt(0);
    __syncthreads();
    if (wave == 0 && it + 1 < nit)
      tdm_load_2d(lds_addr_of(&Bt[cur ^ 1][0]),
                  W + (size_t)n0 * INDIM + (it + 1) * 32,
                  INDIM, EMB, 32, 64, INDIM);
#else
    __syncthreads();
    for (int t = threadIdx.x; t < 64 * 32; t += 256)
      Bt[cur][t] = W[(size_t)(n0 + (t >> 5)) * INDIM + k0 + (t & 31)];
    __syncthreads();
#endif
    v16bf a = a_frag_bf16(arow + k0, kh);
#pragma unroll
    for (int j = 0; j < 4; ++j) {
      v16bf bf = b_frag_f32(&Bt[cur][(j * 16 + ln) * 32], kh);
      acc[j] = wmma_bf16(a, bf, acc[j]);
    }
  }
#pragma unroll
  for (int j = 0; j < 4; ++j) {
    int n = n0 + j * 16 + ln;
    float bias = b[n];
#pragma unroll
    for (int v = 0; v < 8; ++v) {
      size_t m = m0 + kh * 8 + v;
      z[m * EMB + n] = acc[j][v] + bias;
    }
  }
}

__global__ void zero_loss(float* p) {
  if (threadIdx.x == 0 && blockIdx.x == 0) *p = 0.f;
}

// ---------------- residual VQ: 3 codebooks, argmin via WMMA scores ----------
__global__ __launch_bounds__(128) void rvq(const float* __restrict__ z,
                                           const float* __restrict__ E0,
                                           const float* __restrict__ E1,
                                           const float* __restrict__ E2,
                                           float* __restrict__ q,
                                           float* __restrict__ out) {
  __shared__ float res[4][16][EMB];        // 32 KB residual (f32, exact)
  __shared__ float scorebuf[4][16][16];    //  4 KB
  __shared__ int   idxbuf[4][16];
  __shared__ float Enorm[512];             //  2 KB
  __shared__ float lossacc;
  const int lane = threadIdx.x & 31;
  const int wave = threadIdx.x >> 5;
  const int kh   = lane >> 4;
  const int ln   = lane & 15;
  const size_t row0 = (size_t)blockIdx.x * 64;
  const size_t m0   = row0 + wave * 16;

  if (threadIdx.x == 0) lossacc = 0.f;
  for (int t = threadIdx.x; t < 64 * EMB; t += 128) {
    int r = t >> 7, c = t & (EMB - 1);
    res[r >> 4][r & 15][c] = z[(row0 + r) * EMB + c];
  }
  __syncthreads();

  float lsum = 0.f;
  const float* Es[3] = {E0, E1, E2};
  const int    Ks[3] = {8, 64, 512};
  float* codes = out + (size_t)NROWS * EMB + 1;

  for (int s = 0; s < 3; ++s) {
    const float* E = Es[s];
    const int    K = Ks[s];
    for (int j = threadIdx.x; j < K; j += 128) {       // |E_j|^2
      float sm = 0.f;
      for (int c = 0; c < EMB; ++c) { float e = E[(size_t)j * EMB + c]; sm += e * e; }
      Enorm[j] = sm;
    }
    __syncthreads();

    v16bf afr[4];                                       // reuse A across chunks
    const float* ap = &res[wave][ln][0];
#pragma unroll
    for (int ks = 0; ks < 4; ++ks) afr[ks] = a_frag_f32(ap + ks * 32, kh);

    float best = 3.4e38f;
    int   bidx = 0;
    const int nch = (K + 15) / 16;
    for (int ch = 0; ch < nch; ++ch) {
      v8f acc = {};
#pragma unroll
      for (int ks = 0; ks < 4; ++ks) {
        int er = ch * 16 + ln; if (er >= K) er = K - 1;
        v16bf bf = b_frag_f32(E + (size_t)er * EMB + ks * 32, kh);
        acc = wmma_bf16(afr[ks], bf, acc);
      }
      int eg = ch * 16 + ln; if (eg >= K) eg = K - 1;
      float en = Enorm[eg];
#pragma unroll
      for (int v = 0; v < 8; ++v)                      // score = |E|^2 - 2 r.E
        scorebuf[wave][kh * 8 + v][ln] = en - 2.f * acc[v];
      __syncthreads();
      if (lane < 16) {                                  // lane = row, scan 16 cols
        for (int j2 = 0; j2 < 16; ++j2) {
          int gi = ch * 16 + j2;
          float v2 = scorebuf[wave][lane][j2];
          if (gi < K && v2 < best) { best = v2; bidx = gi; }
        }
      }
      __syncthreads();
    }
    if (lane < 16) {
      idxbuf[wave][lane] = bidx;
      codes[(m0 + lane) * 3 + s] = (float)bidx;
    }
    __syncthreads();
    for (int r = 0; r < 16; ++r) {                      // residual -= E[idx]; loss
      int ei = idxbuf[wave][r];
      for (int c = lane; c < EMB; c += 32) {
        float nv = res[wave][r][c] - E[(size_t)ei * EMB + c];
        res[wave][r][c] = nv;
        lsum += nv * nv;
      }
    }
    __syncthreads();
  }

  for (int t = threadIdx.x; t < 64 * EMB; t += 128) {   // q = z - final residual
    int r = t >> 7, c = t & (EMB - 1);
    size_t g = (row0 + r) * EMB + c;
    q[g] = z[g] - res[r >> 4][r & 15][c];
  }
  atomicAdd(&lossacc, lsum);                            // ds_add_f32
  __syncthreads();
  if (threadIdx.x == 0)
    atomicAdd(out + (size_t)NROWS * EMB,
              lossacc * (0.25f / ((float)NROWS * (float)EMB)));
}

// ---------------- fused decoder: out = relu(q W_d1^T + b1) W_d2^T + b2 ------
__global__ __launch_bounds__(128) void decoder(const float* __restrict__ qin,
                                               const float* __restrict__ W1,
                                               const float* __restrict__ b1,
                                               const float* __restrict__ W2,
                                               const float* __restrict__ b2,
                                               float* __restrict__ out) {
  __shared__ unsigned short Wl[EMB * EMB];   // 32 KB, reused W1 then W2
  __shared__ unsigned short h2[4][16 * EMB]; // 16 KB bf16 hidden tile
  __shared__ float bias1[EMB], bias2[EMB];
  const int lane = threadIdx.x & 31;
  const int wave = threadIdx.x >> 5;
  const int kh   = lane >> 4;
  const int ln   = lane & 15;
  const size_t m0 = (size_t)blockIdx.x * 64 + wave * 16;

  for (int t = threadIdx.x; t < EMB * EMB; t += 128) Wl[t] = f2bf(W1[t]);
  for (int t = threadIdx.x; t < EMB; t += 128) { bias1[t] = b1[t]; bias2[t] = b2[t]; }
  __syncthreads();

  v16bf afr[4];
#pragma unroll
  for (int ks = 0; ks < 4; ++ks)
    afr[ks] = a_frag_f32(qin + (m0 + ln) * EMB + ks * 32, kh);

#pragma unroll
  for (int ch = 0; ch < 8; ++ch) {
    v8f acc = {};
#pragma unroll
    for (int ks = 0; ks < 4; ++ks) {
      v16bf bf = b_frag_bf16(Wl + (ch * 16 + ln) * EMB + ks * 32, kh);
      acc = wmma_bf16(afr[ks], bf, acc);
    }
    int n = ch * 16 + ln;
    float bias = bias1[n];
#pragma unroll
    for (int v = 0; v < 8; ++v) {
      float val = acc[v] + bias;
      h2[wave][(kh * 8 + v) * EMB + n] = f2bf(val > 0.f ? val : 0.f);
    }
  }
  __syncthreads();
  for (int t = threadIdx.x; t < EMB * EMB; t += 128) Wl[t] = f2bf(W2[t]);
  __syncthreads();

  v16bf hfr[4];
#pragma unroll
  for (int ks = 0; ks < 4; ++ks)
    hfr[ks] = a_frag_bf16(&h2[wave][ln * EMB + ks * 32], kh);

#pragma unroll
  for (int ch = 0; ch < 8; ++ch) {
    v8f acc = {};
#pragma unroll
    for (int ks = 0; ks < 4; ++ks) {
      v16bf bf = b_frag_bf16(Wl + (ch * 16 + ln) * EMB + ks * 32, kh);
      acc = wmma_bf16(hfr[ks], bf, acc);
    }
    int n = ch * 16 + ln;
    float bias = bias2[n];
#pragma unroll
    for (int v = 0; v < 8; ++v)
      out[(m0 + kh * 8 + v) * EMB + n] = acc[v] + bias;
  }
}

extern "C" void kernel_launch(void* const* d_in, const int* in_sizes, int n_in,
                              void* d_out, int out_size, void* d_ws, size_t ws_size,
                              hipStream_t stream) {
  (void)in_sizes; (void)n_in; (void)out_size; (void)ws_size;
  const float* x   = (const float*)d_in[0];
  const float* We1 = (const float*)d_in[1];
  const float* be1 = (const float*)d_in[2];
  const float* We2 = (const float*)d_in[3];
  const float* be2 = (const float*)d_in[4];
  const float* Wd1 = (const float*)d_in[5];
  const float* bd1 = (const float*)d_in[6];
  const float* Wd2 = (const float*)d_in[7];
  const float* bd2 = (const float*)d_in[8];
  const float* E0  = (const float*)d_in[9];
  const float* E1  = (const float*)d_in[10];
  const float* E2  = (const float*)d_in[11];
  float* out = (float*)d_out;

  // workspace layout: h bf16 [N,768] | z f32 [N,128] | q f32 [N,128]  (~160 MB)
  unsigned short* h = (unsigned short*)d_ws;
  float* z = (float*)((char*)d_ws + (size_t)NROWS * INDIM * sizeof(unsigned short));
  float* q = z + (size_t)NROWS * EMB;

  enc_gemm1<<<dim3(NROWS / 128, INDIM / 64), dim3(256), 0, stream>>>(x, We1, be1, h);
  enc_gemm2<<<dim3(NROWS / 128, EMB / 64),   dim3(256), 0, stream>>>(h, We2, be2, z);
  zero_loss<<<1, 64, 0, stream>>>(out + (size_t)NROWS * EMB);
  rvq<<<dim3(NROWS / 64), dim3(128), 0, stream>>>(z, E0, E1, E2, q, out);
  decoder<<<dim3(NROWS / 64), dim3(128), 0, stream>>>(q, Wd1, bd1, Wd2, bd2, out);
}
